// MultiHeadAttention_15333033247334
// MI455X (gfx1250) — matrix operand order repaired and checked by reading, not compiled, not run on hardware
//
#include <hip/hip_runtime.h>
#include <hip/hip_bf16.h>

// ---- problem constants (fixed by the reference) ----
#define E_DIM   1024
#define S_LEN   2048
#define B_SZ    2
#define H_NUM   16
#define D_HEAD  64
#define M_ROWS  (B_SZ * S_LEN)   // 4096

typedef _Float16 v8h  __attribute__((ext_vector_type(8)));
typedef _Float16 v16h __attribute__((ext_vector_type(16)));
typedef float    v8f  __attribute__((ext_vector_type(8)));

// ---- gfx1250 async global->LDS path (ASYNCcnt), via inline asm ----
// GV mode: LDS address taken from the vdst VGPR; global addr from 64-bit VGPR pair.
// LDS offset = low 32 bits of the generic pointer (aperture is in [63:32]).
static __device__ inline void async_ldg_b128(const _Float16* g, _Float16* l) {
  unsigned           lds = (unsigned)(uintptr_t)l;
  unsigned long long ga  = (unsigned long long)(uintptr_t)g;
  asm volatile("global_load_async_to_lds_b128 %0, %1, off"
               :
               : "v"(lds), "v"(ga)
               : "memory");
}
#define WAIT_ASYNC(n) asm volatile("s_wait_asynccnt " #n ::: "memory")

static __device__ inline v16h cat8(v8h lo, v8h hi) {
  return __builtin_shufflevector(lo, hi, 0,1,2,3,4,5,6,7,8,9,10,11,12,13,14,15);
}

// A fragment (16x32 f16), wave32: lane holds row M = lane%16;
// element j -> K = (lane/16)*16 + j  => two 16B loads per lane.
static __device__ inline v16h load_a_frag(const _Float16* rowp, int halfid) {
  v8h lo = *(const v8h*)(rowp + halfid * 16);
  v8h hi = *(const v8h*)(rowp + halfid * 16 + 8);
  return cat8(lo, hi);
}

// B fragment (32x16 f16): lane holds col N = lane%16, K = (lane/16)*16 + j.
static __device__ inline v16h load_b_frag(const _Float16* p) {
  v8h lo = *(const v8h*)p;
  v8h hi = *(const v8h*)(p + 8);
  return cat8(lo, hi);
}

static __device__ inline v8f wmma_f16f32(v16h a, v16h b, v8f c) {
  return __builtin_amdgcn_wmma_f32_16x16x32_f16(false, a, false, b, (short)0, c,
                                                false, false);
}

// ---------------- conversion kernels ----------------
__global__ void cvt_f32_to_f16(const float* __restrict__ in,
                               _Float16* __restrict__ out, int n) {
  int i = blockIdx.x * blockDim.x + threadIdx.x;
  if (i < n) out[i] = (_Float16)in[i];
}

// W[E][E] (row k, col n) -> WT[n][k] in f16, tiled through LDS
__global__ __launch_bounds__(256) void transpose_cvt(const float* __restrict__ W,
                                                     _Float16* __restrict__ WT) {
  __shared__ float tile[16][17];
  int n = blockIdx.x * 16 + threadIdx.x;
  int k = blockIdx.y * 16 + threadIdx.y;
  tile[threadIdx.y][threadIdx.x] = W[(size_t)k * E_DIM + n];
  __syncthreads();
  int on = blockIdx.x * 16 + threadIdx.y;
  int ok = blockIdx.y * 16 + threadIdx.x;
  WT[(size_t)on * E_DIM + ok] = (_Float16)tile[threadIdx.x][threadIdx.y];
}

// ---------------- shared GEMM mainloop ----------------
// Block = 128 threads (4 waves). Block tile: 128(M) x 64(N).
// Wave tile: 32(M) x 64(N) -> 8 WMMAs per 32-deep k-step.
// B tile (64 rows x 32 k f16) staged to LDS via async copies, double-buffered,
// software-pipelined (branch-free body). LDS rows padded to 40 halves (80B)
// -> conflict-free b128 reads across the 64 banks.
#define BROW 40

// branch-free: thread t copies row (t>>2) and row (t>>2)+32, chunk (t&3)
static __device__ inline void stage_b(const _Float16* __restrict__ WT, int n0,
                                      int k0, _Float16* buf) {
  int t   = (int)threadIdx.x;  // 0..127
  int row = t >> 2, ch = t & 3;
  const _Float16* g = WT + (size_t)(n0 + row) * E_DIM + k0 + ch * 8;
  _Float16*       l = buf + row * BROW + ch * 8;
  async_ldg_b128(g, l);
  async_ldg_b128(g + (size_t)32 * E_DIM, l + 32 * BROW);
}

static __device__ inline void kstep(const _Float16* arow0,
                                    const _Float16* arow1, const _Float16* cur,
                                    int k, int halfid, int nlan,
                                    v8f acc[2][4]) {
  v16h a0 = load_a_frag(arow0 + k, halfid);
  v16h a1 = load_a_frag(arow1 + k, halfid);
#pragma unroll
  for (int n = 0; n < 4; ++n) {
    v16h b = load_b_frag(cur + (n * 16 + nlan) * BROW + halfid * 16);
    acc[0][n] = wmma_f16f32(a0, b, acc[0][n]);
    acc[1][n] = wmma_f16f32(a1, b, acc[1][n]);
  }
}

static __device__ inline void gemm_mainloop(const _Float16* __restrict__ Ah,
                                            const _Float16* __restrict__ WT,
                                            int m0, int n0, _Float16* lds0,
                                            _Float16* lds1, v8f acc[2][4]) {
  const int lane   = threadIdx.x & 31;
  const int halfid = lane >> 4;
  const int nlan   = lane & 15;
  const _Float16* arow0 = Ah + (size_t)(m0 + nlan) * E_DIM;
  const _Float16* arow1 = Ah + (size_t)(m0 + 16 + nlan) * E_DIM;

  stage_b(WT, n0, 0, lds0);
#pragma unroll 1
  for (int k = 0; k < E_DIM - 64; k += 64) {
    stage_b(WT, n0, k + 32, lds1);   // overlap with compute on lds0
    WAIT_ASYNC(2);                   // lds0 landed; lds1 in flight
    __syncthreads();
    kstep(arow0, arow1, lds0, k, halfid, nlan, acc);
    __syncthreads();                 // all waves done reading lds0
    stage_b(WT, n0, k + 64, lds0);
    WAIT_ASYNC(2);
    __syncthreads();
    kstep(arow0, arow1, lds1, k + 32, halfid, nlan, acc);
    __syncthreads();
  }
  // peeled tail: k = E_DIM-64 and k = E_DIM-32
  stage_b(WT, n0, E_DIM - 32, lds1);
  WAIT_ASYNC(2);
  __syncthreads();
  kstep(arow0, arow1, lds0, E_DIM - 64, halfid, nlan, acc);
  WAIT_ASYNC(0);
  __syncthreads();
  kstep(arow0, arow1, lds1, E_DIM - 32, halfid, nlan, acc);
}

// ---------------- QKV projection GEMM ----------------
// grid = (E/64, M/128, 3); block = 128.
__global__ __launch_bounds__(128) void qkv_gemm(
    const _Float16* __restrict__ xh,
    const _Float16* __restrict__ WqT, const _Float16* __restrict__ WkT,
    const _Float16* __restrict__ WvT,
    const float* __restrict__ bq, const float* __restrict__ bk,
    const float* __restrict__ bv,
    _Float16* __restrict__ Qh, _Float16* __restrict__ Kh,
    _Float16* __restrict__ VT) {
  __shared__ _Float16 bstage[2][64 * BROW];
  const int lane   = threadIdx.x & 31;
  const int wave   = threadIdx.x >> 5;
  const int halfid = lane >> 4;
  const int nlan   = lane & 15;
  const int n0 = blockIdx.x * 64;
  const int m0 = blockIdx.y * 128 + wave * 32;
  const int z  = blockIdx.z;

  const _Float16* WT   = (z == 0) ? WqT : ((z == 1) ? WkT : WvT);
  const float*    bias = (z == 0) ? bq  : ((z == 1) ? bk  : bv);

  v8f acc[2][4] = {};
  gemm_mainloop(xh, WT, m0, n0, bstage[0], bstage[1], acc);

#pragma unroll
  for (int n = 0; n < 4; ++n) {
    int col = n0 + n * 16 + nlan;
    int h = col >> 6, d = col & 63;
    float bb = bias[col];
#pragma unroll
    for (int sub = 0; sub < 2; ++sub) {
#pragma unroll
      for (int r = 0; r < 8; ++r) {
        int m = m0 + sub * 16 + r + halfid * 8;
        int b = m >> 11, s = m & (S_LEN - 1);
        float val = acc[sub][n][r] + bb;
        if (z == 2) {
          VT[((size_t)(b * H_NUM + h) * D_HEAD + d) * S_LEN + s] = (_Float16)val;
        } else {
          _Float16* dst = (z == 0) ? Qh : Kh;
          dst[((size_t)(b * H_NUM + h) * S_LEN + s) * D_HEAD + d] = (_Float16)val;
        }
      }
    }
  }
}

// ---------------- flash attention ----------------
// One wave handles one (b,h, 16-query) tile; 32 keys per step.
__global__ __launch_bounds__(128) void attn_kernel(
    const _Float16* __restrict__ Qh, const _Float16* __restrict__ Kh,
    const _Float16* __restrict__ VT, _Float16* __restrict__ ctx,
    const int* __restrict__ maskp) {
  __shared__ _Float16 pt_all[4][16 * 32];
  const int wave   = threadIdx.x >> 5;
  const int lane   = threadIdx.x & 31;
  const int halfid = lane >> 4;
  const int nlan   = lane & 15;
  _Float16* pt = pt_all[wave];

  const int qt = blockIdx.x & (S_LEN / 16 - 1);  // same for all waves in block
  const int bh = (blockIdx.x >> 7) * 4 + wave;   // 0..31
  const int q0 = qt * 16;
  const bool causal = (maskp[0] != 0);

  const _Float16* Qb = Qh + (size_t)bh * S_LEN * D_HEAD;
  const _Float16* Kb = Kh + (size_t)bh * S_LEN * D_HEAD;
  const _Float16* Vb = VT + (size_t)bh * D_HEAD * S_LEN;

  const _Float16* qrow = Qb + (size_t)(q0 + nlan) * D_HEAD;
  v16h aq0 = load_a_frag(qrow, halfid);
  v16h aq1 = load_a_frag(qrow + 32, halfid);

  float m_r[8], l_r[8];
  v8f acc[4] = {};
#pragma unroll
  for (int r = 0; r < 8; ++r) { m_r[r] = -1e30f; l_r[r] = 0.0f; }

  const float scale = 0.03125f;  // 1/sqrt(1024)
  const int tmax = causal ? (q0 + 16) : S_LEN;

  for (int t0 = 0; t0 < tmax; t0 += 32) {
    // scores: two 16x16 tiles over 32 keys
    v8f sc0 = {}, sc1 = {};
    {
      const _Float16* kr0 = Kb + (size_t)(t0 + nlan) * D_HEAD + halfid * 16;
      sc0 = wmma_f16f32(aq0, load_b_frag(kr0), sc0);
      sc0 = wmma_f16f32(aq1, load_b_frag(kr0 + 32), sc0);
      const _Float16* kr1 = Kb + (size_t)(t0 + 16 + nlan) * D_HEAD + halfid * 16;
      sc1 = wmma_f16f32(aq0, load_b_frag(kr1), sc1);
      sc1 = wmma_f16f32(aq1, load_b_frag(kr1 + 32), sc1);
    }
    // V fragments issued early so they overlap the softmax VALU work
    v16h vbf[4];
#pragma unroll
    for (int n = 0; n < 4; ++n)
      vbf[n] =
          load_b_frag(Vb + (size_t)(n * 16 + nlan) * S_LEN + t0 + halfid * 16);
    // online softmax across the 32 columns
#pragma unroll
    for (int r = 0; r < 8; ++r) {
      int q   = q0 + r + halfid * 8;
      int tc0 = t0 + nlan;
      int tc1 = t0 + 16 + nlan;
      float s0 = (!causal || tc0 <= q) ? sc0[r] * scale : -1e9f;
      float s1 = (!causal || tc1 <= q) ? sc1[r] * scale : -1e9f;
      float rmax = fmaxf(s0, s1);
#pragma unroll
      for (int off = 1; off < 16; off <<= 1)
        rmax = fmaxf(rmax, __shfl_xor(rmax, off, 32));
      float mnew  = fmaxf(m_r[r], rmax);
      float alpha = __expf(m_r[r] - mnew);
      float p0 = __expf(s0 - mnew);
      float p1 = __expf(s1 - mnew);
      float rs = p0 + p1;
#pragma unroll
      for (int off = 1; off < 16; off <<= 1) rs += __shfl_xor(rs, off, 32);
      l_r[r] = l_r[r] * alpha + rs;
      m_r[r] = mnew;
      int row = r + halfid * 8;
      pt[row * 32 + nlan]      = (_Float16)p0;
      pt[row * 32 + 16 + nlan] = (_Float16)p1;
#pragma unroll
      for (int n = 0; n < 4; ++n) acc[n][r] *= alpha;
    }
    // DS ops are in-order per wave; drain before the transposed reload.
    asm volatile("s_wait_dscnt 0x0" ::: "memory");
    v16h pa = load_a_frag(pt + nlan * 32, halfid);
#pragma unroll
    for (int n = 0; n < 4; ++n) acc[n] = wmma_f16f32(pa, vbf[n], acc[n]);
  }

  const int b = bh >> 4, h = bh & 15;
#pragma unroll
  for (int r = 0; r < 8; ++r) {
    int q = q0 + r + halfid * 8;
    float inv = 1.0f / l_r[r];
#pragma unroll
    for (int n = 0; n < 4; ++n) {
      int d = n * 16 + nlan;
      ctx[((size_t)(b * S_LEN + q)) * E_DIM + h * D_HEAD + d] =
          (_Float16)(acc[n][r] * inv);
    }
  }
}

// ---------------- output projection: out = ctx @ Wo + bo (fp32) -------------
__global__ __launch_bounds__(128) void out_gemm(
    const _Float16* __restrict__ ctx, const _Float16* __restrict__ WoT,
    const float* __restrict__ bo, float* __restrict__ out) {
  __shared__ _Float16 bstage[2][64 * BROW];
  const int lane   = threadIdx.x & 31;
  const int wave   = threadIdx.x >> 5;
  const int halfid = lane >> 4;
  const int nlan   = lane & 15;
  const int n0 = blockIdx.x * 64;
  const int m0 = blockIdx.y * 128 + wave * 32;

  v8f acc[2][4] = {};
  gemm_mainloop(ctx, WoT, m0, n0, bstage[0], bstage[1], acc);

#pragma unroll
  for (int n = 0; n < 4; ++n) {
    int col = n0 + n * 16 + nlan;
    float bb = bo[col];
#pragma unroll
    for (int sub = 0; sub < 2; ++sub) {
#pragma unroll
      for (int r = 0; r < 8; ++r) {
        int m = m0 + sub * 16 + r + halfid * 8;
        out[(size_t)m * E_DIM + col] = acc[sub][n][r] + bb;
      }
    }
  }
}

// ---------------- host launcher ----------------
extern "C" void kernel_launch(void* const* d_in, const int* in_sizes, int n_in,
                              void* d_out, int out_size, void* d_ws,
                              size_t ws_size, hipStream_t stream) {
  const float* x    = (const float*)d_in[0];
  const int*   mask = (const int*)d_in[1];
  const float* Wq   = (const float*)d_in[2];
  const float* bq   = (const float*)d_in[3];
  const float* Wk   = (const float*)d_in[4];
  const float* bk   = (const float*)d_in[5];
  const float* Wv   = (const float*)d_in[6];
  const float* bv   = (const float*)d_in[7];
  const float* Wo   = (const float*)d_in[8];
  const float* bo   = (const float*)d_in[9];
  float* out = (float*)d_out;

  const size_t XH_E = (size_t)M_ROWS * E_DIM;
  const size_t WT_E = (size_t)E_DIM * E_DIM;
  const size_t QV_E = (size_t)B_SZ * H_NUM * S_LEN * D_HEAD;
  _Float16* xh  = (_Float16*)d_ws;
  _Float16* WqT = xh + XH_E;
  _Float16* WkT = WqT + WT_E;
  _Float16* WvT = WkT + WT_E;
  _Float16* WoT = WvT + WT_E;
  _Float16* Qh  = WoT + WT_E;
  _Float16* Kh  = Qh + QV_E;
  _Float16* VT  = Kh + QV_E;
  _Float16* ctx = VT + QV_E;

  {
    int n = (int)XH_E;
    cvt_f32_to_f16<<<(n + 255) / 256, 256, 0, stream>>>(x, xh, n);
  }
  {
    dim3 g(E_DIM / 16, E_DIM / 16), b(16, 16);
    transpose_cvt<<<g, b, 0, stream>>>(Wq, WqT);
    transpose_cvt<<<g, b, 0, stream>>>(Wk, WkT);
    transpose_cvt<<<g, b, 0, stream>>>(Wv, WvT);
    transpose_cvt<<<g, b, 0, stream>>>(Wo, WoT);
  }
  {
    dim3 g(E_DIM / 64, M_ROWS / 128, 3);
    qkv_gemm<<<g, 128, 0, stream>>>(xh, WqT, WkT, WvT, bq, bk, bv, Qh, Kh, VT);
  }
  {
    dim3 g((B_SZ * H_NUM / 4) * (S_LEN / 16));
    attn_kernel<<<g, 128, 0, stream>>>(Qh, Kh, VT, ctx, mask);
  }
  {
    dim3 g(E_DIM / 64, M_ROWS / 128);
    out_gemm<<<g, 128, 0, stream>>>(ctx, WoT, bo, out);
  }
}
